// QuantizedLinear_75024488726593
// MI455X (gfx1250) — matrix operand order repaired and checked by reading, not compiled, not run on hardware
//
#include <hip/hip_runtime.h>
#include <hip/hip_fp16.h>
#include <stdint.h>

// W4A16 grouped-quant GEMM for MI455X (gfx1250), wave32 + WMMA.
// x:[M,K] f16, qweight:[N,K/8] i32 (4-bit, low nibble = lowest K),
// scale/zero:[N,K/128] f16, bias:[N] f16 -> out:[M,N] f16.

typedef __attribute__((ext_vector_type(16))) _Float16 v16h;
typedef __attribute__((ext_vector_type(8)))  _Float16 v8h;
typedef __attribute__((ext_vector_type(2)))  _Float16 h2;
typedef __attribute__((ext_vector_type(8)))  float    v8f;

#define WAVES_PER_BLOCK 8
#define GROUP_SIZE 128

__global__ __launch_bounds__(WAVES_PER_BLOCK * 32)
void wq4_linear_wmma(const _Float16* __restrict__ x,
                     const int32_t*  __restrict__ qweight,
                     const _Float16* __restrict__ scale,
                     const _Float16* __restrict__ zero_point,
                     const _Float16* __restrict__ bias,
                     _Float16*       __restrict__ out,
                     int M, int K, int N, int num_groups)
{
    const int lane = threadIdx.x & 31;
    const int wave = threadIdx.x >> 5;
    const int tile = blockIdx.x * WAVES_PER_BLOCK + wave;   // 16-wide N tile
    const int n0   = tile * 16;
    if (n0 >= N) return;                                    // wave-uniform exit

    const int col = lane & 15;        // N column within tile / A row index
    const int hi  = lane >> 4;        // 0: K 0..15 of step, 1: K 16..31
    int n = n0 + col;
    const bool ncol_valid = (n < N);
    if (!ncol_valid) n = N - 1;       // clamp loads; stores predicated below

    const int packed_K = K >> 3;      // int32s per weight row
    // Rows >= M feed dummy (row 0) data into A; D rows >= M are never stored.
    const int arow = (col < M) ? col : 0;

    const _Float16* xrow = x          + (size_t)arow * K;
    const int32_t*  qrow = qweight    + (size_t)n * packed_K;
    const _Float16* srow = scale      + (size_t)n * num_groups;
    const _Float16* zrow = zero_point + (size_t)n * num_groups;

    v8f acc = {};   // 16x16 f32 accumulator fragment (8 VGPRs)

    for (int g = 0; g < num_groups; ++g) {
        const float sv = (float)srow[g];
        const float zv = (float)zrow[g];
        // w = s*(q - z) = s*(1024+q) + (-s*(1024+z)); 1024+q exact in fp16.
        const _Float16 sh = (_Float16)sv;
        const _Float16 ch = (_Float16)(-sv * (1024.0f + zv));
        const h2 spk = { sh, sh };
        const h2 cpk = { ch, ch };

        const int kg = g << 7;   // GROUP_SIZE = 128 K per group

        // Stream-ahead prefetch of qweight (~2 groups = 128B ahead per lane).
        __builtin_prefetch(qrow + (kg >> 3) + 32, 0, 1);

#pragma unroll
        for (int kk = 0; kk < GROUP_SIZE; kk += 32) {
            const int k0 = kg + kk;

            // ---- A fragment: 16x32 f16. Layout: lanes 0-15 hold
            // K {0..15}, lanes 16-31 hold K {16..31} (matches B).
            union { v16h v; v8h h[2]; } a;
            a.h[0] = *(const v8h*)(xrow + k0 + hi * 16);       // b128
            a.h[1] = *(const v8h*)(xrow + k0 + hi * 16 + 8);   // b128

            // ---- B fragment: 32x16 f16, one N column per lane.
            // Lane needs 16 consecutive nibbles of row n -> one b64 load.
            const int32_t* qp = qrow + (k0 >> 3) + hi * 2;
            const uint32_t q0 = (uint32_t)qp[0];
            const uint32_t q1 = (uint32_t)qp[1];

            union { v16h v; h2 h[8]; } b;
#pragma unroll
            for (int d = 0; d < 2; ++d) {
                const uint32_t q = d ? q1 : q0;
                // Pack nibble pairs into fp16 "1024 + q" lanes (0x6400 trick),
                // then one packed FMA per pair -> v_pk_fma_f16.
                const uint32_t u0 = 0x64006400u | (q & 0xFu)         | ((q << 12) & 0xF0000u);
                const uint32_t u1 = 0x64006400u | ((q >> 8)  & 0xFu) | ((q << 4)  & 0xF0000u);
                const uint32_t u2 = 0x64006400u | ((q >> 16) & 0xFu) | ((q >> 4)  & 0xF0000u);
                const uint32_t u3 = 0x64006400u | ((q >> 24) & 0xFu) | ((q >> 12) & 0xF0000u);
                b.h[d * 4 + 0] = __builtin_bit_cast(h2, u0) * spk + cpk;
                b.h[d * 4 + 1] = __builtin_bit_cast(h2, u1) * spk + cpk;
                b.h[d * 4 + 2] = __builtin_bit_cast(h2, u2) * spk + cpk;
                b.h[d * 4 + 3] = __builtin_bit_cast(h2, u3) * spk + cpk;
            }

            // EXEC is all-ones here (only wave-uniform early exit above).
            acc = __builtin_amdgcn_wmma_f32_16x16x32_f16(
                /*neg_a=*/false, a.v, /*neg_b=*/false, b.v,
                /*c_mod=*/(short)0, acc, /*reuse_a=*/false, /*reuse_b=*/false);
        }
    }

    // ---- Epilogue: C/D layout puts M = hi*8 + r in VGPR r of this half.
    if (ncol_valid) {
        const float bv = (float)bias[n];
        const int mbase = hi * 8;
#pragma unroll
        for (int r = 0; r < 8; ++r) {
            const int m = mbase + r;
            if (m < M)
                out[(size_t)m * N + n] = (_Float16)(acc[r] + bv);
        }
    }
}

extern "C" void kernel_launch(void* const* d_in, const int* in_sizes, int n_in,
                              void* d_out, int out_size, void* d_ws, size_t ws_size,
                              hipStream_t stream) {
    const _Float16* x    = (const _Float16*)d_in[0];
    const int32_t*  qw   = (const int32_t*) d_in[1];
    const _Float16* sc   = (const _Float16*)d_in[2];
    const _Float16* zp   = (const _Float16*)d_in[3];
    const _Float16* bs   = (const _Float16*)d_in[4];
    _Float16*       outp = (_Float16*)d_out;

    const int N          = in_sizes[4];                 // bias length
    const int num_groups = in_sizes[2] / N;             // scale[N, G]
    const int K          = num_groups * GROUP_SIZE;     // 4096
    const int M          = in_sizes[0] / K;             // 8

    const int tiles  = (N + 15) / 16;                   // 896
    const int blocks = (tiles + WAVES_PER_BLOCK - 1) / WAVES_PER_BLOCK;

    wq4_linear_wmma<<<blocks, WAVES_PER_BLOCK * 32, 0, stream>>>(
        x, qw, sc, zp, bs, outp, M, K, N, num_groups);
}
